// BLA_51410758533352
// MI455X (gfx1250) — matrix operand order repaired and checked by reading, not compiled, not run on hardware
//
#include <hip/hip_runtime.h>

// ---------------------------------------------------------------------------
// MI455X (gfx1250) BLA forward.
// GEMMs: v_wmma_f32_16x16x32_bf16 (wave32). B tile staged in LDS via
// global_load_async_to_lds_b128 (double-buffered, ASYNCcnt-paced); A fragment
// register double-buffered so both operand streams overlap the WMMA chain.
// Attention / LN / upsample are memory-shaped -> plain VALU wave32 kernels.
// ---------------------------------------------------------------------------

typedef __bf16 bf16;
typedef __attribute__((ext_vector_type(16))) __bf16 bf16x16;
typedef __attribute__((ext_vector_type(8)))  float  f32x8;

__device__ __forceinline__ int iclamp(int v, int lo, int hi) {
  return v < lo ? lo : (v > hi ? hi : v);
}

// issue one 16B global->LDS async transfer (per lane)
__device__ __forceinline__ void async_g2l_b128(const bf16* g, bf16* l) {
  unsigned laddr = (unsigned)(uintptr_t)l;  // LDS byte address (low 32 bits)
  asm volatile("global_load_async_to_lds_b128 %0, %1, off"
               :: "v"(laddr), "v"(g) : "memory");
}

// ---------------------------------------------------------------------------
// C[M,N] = A[M,K](bf16) * B[K,N](bf16, pre-transposed) + bias[N]
// template: ADD -> C += ...; RELU -> max(0); OUTBF16 -> store bf16 into Cb.
// Block = 128 threads = 4 waves; block tile 64x64; each wave 16(M) x 64(N)
// with 4 f32x8 accumulators. B tile (32x64) in LDS, double-buffered; A
// fragment double-buffered in registers (next tile loads during WMMAs).
// ---------------------------------------------------------------------------
#define BROW 72  // LDS row stride in bf16 (64 data + 8 pad -> 144B, 16B aligned)

template <bool ADD, bool RELU, bool OUTBF16>
__global__ __launch_bounds__(128) void gemm_wmma(
    const bf16* __restrict__ A, const bf16* __restrict__ Bm,
    const float* __restrict__ bias,
    float* __restrict__ C, bf16* __restrict__ Cb,
    int M, int N, int K) {
  __shared__ bf16 ldsB[2][32 * BROW];

  const int tid   = threadIdx.x;
  const int lane  = tid & 31;
  const int wave  = tid >> 5;
  const int tileN = blockIdx.x * 64;
  const int tileM = blockIdx.y * 64 + wave * 16;

  // B-tile loader: 4KB tile = 256 x 16B transfers; 128 threads -> 2 each.
  auto issueB = [&](int buf, int k0) {
#pragma unroll
    for (int j = 0; j < 2; ++j) {
      int e   = tid + 128 * j;
      int row = e >> 3;            // K row within tile
      int cb  = (e & 7) << 3;      // bf16 column offset (8 bf16 = 16B chunk)
      const bf16* g = Bm + (size_t)(k0 + row) * N + tileN + cb;
      async_g2l_b128(g, &ldsB[buf][row * BROW + cb]);
    }
  };

  // A fragment mapping (ISA 7.12.2, 16-bit A 16x32):
  //   m = lane&15; lanes 0-15 hold K quads {0..7,16..23}, lanes 16-31 {8..15,24..31}
  const int m  = tileM + (lane & 15);
  const int kq = (lane >> 4) << 3;
  const bf16* ap = A + (size_t)m * K;

  auto loadA = [&](int k0) {
    bf16x16 af;
#pragma unroll
    for (int j = 0; j < 8; ++j) af[j] = ap[k0 + kq + j];
#pragma unroll
    for (int j = 0; j < 8; ++j) af[8 + j] = ap[k0 + 16 + kq + j];
    return af;
  };

  f32x8 acc[4] = {};

  issueB(0, 0);
  bf16x16 af = loadA(0);
  int buf = 0;
  for (int k0 = 0; k0 < K; k0 += 32) {
    const bool more = (k0 + 32 < K);
    if (more) {
      issueB(buf ^ 1, k0 + 32);  // stream next B tile while we compute
      asm volatile("s_wait_asynccnt 0x2" ::: "memory");  // current tile landed
    } else {
      asm volatile("s_wait_asynccnt 0x0" ::: "memory");
    }
    __syncthreads();  // publish LDS tile to all 4 waves

    // pull all 4 B fragments (16 row-strided bf16 LDS reads each) before the WMMA chain
    bf16x16 bfrag[4];
    const bf16* brow = &ldsB[buf][((lane >> 4) << 3) * BROW + (lane & 15)];
#pragma unroll
    for (int nt = 0; nt < 4; ++nt) {
#pragma unroll
      for (int e = 0; e < 16; ++e) bfrag[nt][e] = brow[(e + ((e >> 3) << 3)) * BROW + nt * 16];
    }

    // next A fragment in flight during the WMMAs
    bf16x16 afn;
    if (more) afn = loadA(k0 + 32);

#pragma unroll
    for (int nt = 0; nt < 4; ++nt)
      acc[nt] = __builtin_amdgcn_wmma_f32_16x16x32_bf16(
          /*neg_a=*/false, af, /*neg_b=*/false, bfrag[nt],
          /*c_mod=*/(short)0, acc[nt], /*reuse_a=*/false, /*reuse_b=*/false);

    __syncthreads();  // all waves done reading before next tile overwrites
    if (more) af = afn;
    buf ^= 1;
  }

  // C/D mapping: VGPR r -> row = r + 8*lane[4]; col = lane&15 (+16*nt)
  const int rowBase = tileM + ((lane >> 4) << 3);
  const int col0    = tileN + (lane & 15);
#pragma unroll
  for (int nt = 0; nt < 4; ++nt) {
    const int col = col0 + nt * 16;
    const float bv = bias[col];
#pragma unroll
    for (int r = 0; r < 8; ++r) {
      const size_t off = (size_t)(rowBase + r) * N + col;
      float v = acc[nt][r] + bv;
      if (RELU) v = v > 0.f ? v : 0.f;
      if (ADD) v += C[off];
      if (OUTBF16) Cb[off] = (bf16)v;
      else         C[off]  = v;
    }
  }
}

// ---------------------------------------------------------------------------
__global__ void f32_to_bf16(const float* __restrict__ in, bf16* __restrict__ out, int n) {
  int i = blockIdx.x * blockDim.x + threadIdx.x;
  if (i < n) out[i] = (bf16)in[i];
}

// Bm[k*N + n] = W[n*ldw + k]  (weight transpose + bf16 convert)
__global__ void transpose_to_bf16(const float* __restrict__ W, bf16* __restrict__ Bm,
                                  int K, int N, int ldw) {
  int i = blockIdx.x * blockDim.x + threadIdx.x;
  if (i >= K * N) return;
  int k = i / N, n = i - k * N;
  Bm[i] = (bf16)W[(size_t)n * ldw + k];
}

// ---------------------------------------------------------------------------
// 3x3 local-window attention, head-dim fixed = 32 for every level.
// One thread per (batch, out_y, out_x, head); q pixel = (oy*sq, ox*sq) in the
// q map (qH x qW), window center = (oy*sw, ox*sw) in the source map (sH x sW),
// taps replicate-padded (clamp). k gets sc*k + pos[c*9+t]; q pre-scaled by sc.
// Output stored bf16 (feeds the out-projection WMMA GEMM directly).
// ---------------------------------------------------------------------------
__global__ void local_attn_kernel(
    const float* __restrict__ qb, int qStride, int qOff, int qH, int qW, int sq,
    const float* __restrict__ kb, int kStride, int kOff, int vOff, int sH, int sW, int sw,
    const float* __restrict__ pos, float sc, int nh,
    bf16* __restrict__ out, int f, int oh, int ow, int total) {
  int tid = blockIdx.x * blockDim.x + threadIdx.x;
  if (tid >= total) return;
  int head = tid % nh;
  int pix  = tid / nh;
  int ox = pix % ow;
  int t1 = pix / ow;
  int oy = t1 % oh;
  int bb = t1 / oh;

  const float* qp = qb + (size_t)((bb * qH + oy * sq) * qW + ox * sq) * qStride + qOff + head * 32;
  float q[32];
#pragma unroll
  for (int d = 0; d < 32; ++d) q[d] = sc * qp[d];

  const float* pp = pos + (size_t)head * 32 * 9;
  float lg[9];
  float mx = -3.4e38f;
#pragma unroll
  for (int t = 0; t < 9; ++t) {
    int yy = iclamp(oy * sw + t / 3 - 1, 0, sH - 1);
    int xx = iclamp(ox * sw + t % 3 - 1, 0, sW - 1);
    const float* kp = kb + (size_t)((bb * sH + yy) * sW + xx) * kStride + kOff + head * 32;
    float acc = 0.f;
#pragma unroll
    for (int d = 0; d < 32; ++d) acc += q[d] * (sc * kp[d] + pp[d * 9 + t]);
    lg[t] = acc;
    mx = fmaxf(mx, acc);
  }
  float se = 0.f;
#pragma unroll
  for (int t = 0; t < 9; ++t) { lg[t] = expf(lg[t] - mx); se += lg[t]; }
  float inv = 1.f / se;

  float o[32];
#pragma unroll
  for (int d = 0; d < 32; ++d) o[d] = 0.f;
#pragma unroll
  for (int t = 0; t < 9; ++t) {
    int yy = iclamp(oy * sw + t / 3 - 1, 0, sH - 1);
    int xx = iclamp(ox * sw + t % 3 - 1, 0, sW - 1);
    const float* vp = kb + (size_t)((bb * sH + yy) * sW + xx) * kStride + vOff + head * 32;
    float a = lg[t] * inv;
#pragma unroll
    for (int d = 0; d < 32; ++d) o[d] += a * vp[d];
  }
  bf16* op = out + (size_t)((bb * oh + oy) * ow + ox) * f + head * 32;
#pragma unroll
  for (int d = 0; d < 32; ++d) op[d] = (bf16)o[d];
}

// ---------------------------------------------------------------------------
// y = LN(a + b) * g + beta. One wave32 per row, shfl_xor reductions.
// Optionally writes f32 and/or bf16 copies (bf16 feeds the next WMMA GEMM).
// f in {128,256,512} -> <=16 values per lane.
// ---------------------------------------------------------------------------
__global__ void ln_add_kernel(const float* __restrict__ a, const float* __restrict__ b,
                              const float* __restrict__ g, const float* __restrict__ beta,
                              float* __restrict__ outf, bf16* __restrict__ outb,
                              int rows, int f) {
  int lane = threadIdx.x & 31;
  int wave = threadIdx.x >> 5;
  int row  = blockIdx.x * (blockDim.x >> 5) + wave;
  if (row >= rows) return;
  const float* ap = a + (size_t)row * f;
  const float* bp = b + (size_t)row * f;
  const int n = f >> 5;
  float xv[16];
  float s = 0.f;
  for (int j = 0; j < n; ++j) {
    float v = ap[lane + 32 * j] + bp[lane + 32 * j];
    xv[j] = v;
    s += v;
  }
#pragma unroll
  for (int msk = 16; msk; msk >>= 1) s += __shfl_xor(s, msk, 32);
  const float mu = s / f;
  float vs = 0.f;
  for (int j = 0; j < n; ++j) { float d = xv[j] - mu; vs += d * d; }
#pragma unroll
  for (int msk = 16; msk; msk >>= 1) vs += __shfl_xor(vs, msk, 32);
  const float inv = rsqrtf(vs / f + 1e-5f);
  for (int j = 0; j < n; ++j) {
    int c = lane + 32 * j;
    float y = (xv[j] - mu) * inv * g[c] + beta[c];
    if (outf) outf[(size_t)row * f + c] = y;
    if (outb) outb[(size_t)row * f + c] = (bf16)y;
  }
}

// ---------------------------------------------------------------------------
// dst[b,y,x,c] += bilinear_align_corners(src)[b,y,x,c], H1=2*H2.
// ---------------------------------------------------------------------------
__global__ void upsample_add_kernel(float* __restrict__ dst, const float* __restrict__ src,
                                    int H1, int W1, int H2, int W2, int C,
                                    float ry, float rx, int total) {
  int idx = blockIdx.x * blockDim.x + threadIdx.x;
  if (idx >= total) return;
  int c = idx % C;
  int t = idx / C;
  int x = t % W1; t /= W1;
  int y = t % H1;
  int b = t / H1;
  float fy = y * ry;
  int y0 = (int)fy; if (y0 > H2 - 2) y0 = H2 - 2;
  float wy = fy - y0;
  float fx = x * rx;
  int x0 = (int)fx; if (x0 > W2 - 2) x0 = W2 - 2;
  float wx = fx - x0;
  const float* s = src + (size_t)b * H2 * W2 * C;
  size_t i00 = ((size_t)y0 * W2 + x0) * C + c;
  float v00 = s[i00];
  float v01 = s[i00 + C];
  float v10 = s[i00 + (size_t)W2 * C];
  float v11 = s[i00 + (size_t)W2 * C + C];
  float v = (v00 * (1.f - wx) + v01 * wx) * (1.f - wy) +
            (v10 * (1.f - wx) + v11 * wx) * wy;
  dst[idx] += v;
}

// ---------------------------------------------------------------------------
extern "C" void kernel_launch(void* const* d_in, const int* in_sizes, int n_in,
                              void* d_out, int out_size, void* d_ws, size_t ws_size,
                              hipStream_t stream) {
  (void)in_sizes; (void)n_in; (void)out_size; (void)ws_size;

  const int HH[3]    = {128, 64, 32};
  const int F[3]     = {128, 256, 512};
  const int NHh[3]   = {4, 8, 16};
  const int PSZ[3]   = {1024, 2560, 2560};
  const int Mrows[3] = {2 * 128 * 128, 2 * 64 * 64, 2 * 32 * 32};
  const float SC = 0.42044820762685725f;  // 32^-0.25 (head dim 32 everywhere)

  const int fIdx[3]  = {0, 14, 28};
  const int pBase[3] = {1, 15, 29};
  enum { P_INW = 0, P_INB, P_POS, P_OUTW, P_OUTB, P_LN1G, P_LN1B,
         P_W1, P_B1, P_W2, P_B2, P_LN2G, P_LN2B };
  auto fin = [&](int i) -> const float* { return (const float*)d_in[i]; };

  // bump allocator over d_ws
  char* wsBase = (char*)d_ws;
  size_t wsOff = 0;
  auto alloc = [&](size_t bytes) -> void* {
    void* p = wsBase + wsOff;
    wsOff = (wsOff + bytes + 255) & ~(size_t)255;
    return p;
  };

  bf16 *featb[3], *inwT[3], *outwT[3], *w1T[3], *w2T[3], *wself[3], *xlnb[3], *hffb[3];
  bf16 *wtd[2], *wbu[2];
  float *proj[3], *delta[3], *xlnf[3], *y2[3], *tdtmp[2];

  for (int i = 0; i < 3; ++i) {
    size_t MF = (size_t)Mrows[i] * F[i];
    featb[i] = (bf16*)alloc(MF * 2);
    inwT[i]  = (bf16*)alloc((size_t)F[i] * PSZ[i] * 2);
    outwT[i] = (bf16*)alloc((size_t)3 * F[i] * F[i] * 2);
    w1T[i]   = (bf16*)alloc((size_t)F[i] * 4 * F[i] * 2);
    w2T[i]   = (bf16*)alloc((size_t)4 * F[i] * F[i] * 2);
    wself[i] = (bf16*)alloc(MF * 2);
    xlnb[i]  = (bf16*)alloc(MF * 2);
    hffb[i]  = (bf16*)alloc(MF * 4 * 2);
    proj[i]  = (float*)alloc((size_t)Mrows[i] * PSZ[i] * 4);
    delta[i] = (float*)alloc(MF * 4);
    xlnf[i]  = (float*)alloc(MF * 4);
    y2[i]    = (float*)alloc(MF * 4);
  }
  for (int i = 0; i < 2; ++i) {
    size_t MT = (size_t)(Mrows[i] / 4) * F[i];
    wtd[i]   = (bf16*)alloc(MT * 2);
    tdtmp[i] = (float*)alloc(MT * 4);
    wbu[i]   = (bf16*)alloc((size_t)Mrows[i + 1] * F[i + 1] * 2);
  }

  auto gemm = [&](const bf16* A, const bf16* Bm, const float* bias, float* C, bf16* Cb,
                  int Mi, int Ni, int Ki, int mode) {
    dim3 grid(Ni / 64, Mi / 64), blk(128);
    if (mode == 0)
      gemm_wmma<false, false, false><<<grid, blk, 0, stream>>>(A, Bm, bias, C, Cb, Mi, Ni, Ki);
    else if (mode == 1)  // accumulate into C
      gemm_wmma<true, false, false><<<grid, blk, 0, stream>>>(A, Bm, bias, C, Cb, Mi, Ni, Ki);
    else                 // relu + bf16 out
      gemm_wmma<false, true, true><<<grid, blk, 0, stream>>>(A, Bm, bias, C, Cb, Mi, Ni, Ki);
  };

  auto attn = [&](const float* qb, int qs, int qoff, int qH, int qW, int sq,
                  const float* kb, int ks, int koff, int voff, int sH, int sW, int sw,
                  const float* pos, int nh, int f, bf16* out, int oh, int ow) {
    int total = 2 * oh * ow * nh;
    local_attn_kernel<<<(total + 255) / 256, 256, 0, stream>>>(
        qb, qs, qoff, qH, qW, sq, kb, ks, koff, voff, sH, sW, sw,
        pos, SC, nh, out, f, oh, ow, total);
  };

  // ---- prep: bf16 features + transposed bf16 weights --------------------
  for (int i = 0; i < 3; ++i) {
    int n = Mrows[i] * F[i];
    f32_to_bf16<<<(n + 255) / 256, 256, 0, stream>>>(fin(fIdx[i]), featb[i], n);
    int t;
    t = F[i] * PSZ[i];
    transpose_to_bf16<<<(t + 255) / 256, 256, 0, stream>>>(fin(pBase[i] + P_INW), inwT[i], F[i], PSZ[i], F[i]);
    t = 3 * F[i] * F[i];
    transpose_to_bf16<<<(t + 255) / 256, 256, 0, stream>>>(fin(pBase[i] + P_OUTW), outwT[i], 3 * F[i], F[i], 3 * F[i]);
    t = F[i] * 4 * F[i];
    transpose_to_bf16<<<(t + 255) / 256, 256, 0, stream>>>(fin(pBase[i] + P_W1), w1T[i], F[i], 4 * F[i], F[i]);
    transpose_to_bf16<<<(t + 255) / 256, 256, 0, stream>>>(fin(pBase[i] + P_W2), w2T[i], 4 * F[i], F[i], 4 * F[i]);
  }

  // ---- input projections (WMMA) -----------------------------------------
  for (int i = 0; i < 3; ++i)
    gemm(featb[i], inwT[i], fin(pBase[i] + P_INB), proj[i], nullptr,
         Mrows[i], PSZ[i], F[i], 0);

  const float* pos0 = fin(pBase[0] + P_POS);
  const float* pos1 = fin(pBase[1] + P_POS);
  const float* pos2 = fin(pBase[2] + P_POS);

  // ---- attention --------------------------------------------------------
  // self (stride-1 windows on own map)
  for (int i = 0; i < 3; ++i)
    attn(proj[i], PSZ[i], 0, HH[i], HH[i], 1,
         proj[i], PSZ[i], F[i], 2 * F[i], HH[i], HH[i], 1,
         fin(pBase[i] + P_POS), NHh[i], F[i], wself[i], HH[i], HH[i]);
  // top-down: level-i queries (::2 subsample) over level-(i+1) windows
  attn(proj[0], 1024, 3 * 128, 128, 128, 2,
       proj[1], 2560, 1024, 1152, 64, 64, 1,
       pos0 + 128 * 9, 4, 128, wtd[0], 64, 64);
  attn(proj[1], 2560, 3 * 256, 64, 64, 2,
       proj[2], 2560, 1536, 1792, 32, 32, 1,
       pos1 + 256 * 9, 8, 256, wtd[1], 32, 32);
  // bottom-up: level-(i+1) queries over stride-2 windows of level-i
  attn(proj[1], 2560, 1280, 64, 64, 1,
       proj[0], 1024, 512, 768, 128, 128, 2,
       pos1 + 512 * 9, 8, 256, wbu[0], 64, 64);
  attn(proj[2], 2560, 2048, 32, 32, 1,
       proj[1], 2560, 1536, 2048, 64, 64, 2,
       pos2 + 512 * 9, 16, 512, wbu[1], 32, 32);

  // ---- output projections -> deltas (WMMA) ------------------------------
  for (int i = 0; i < 3; ++i)
    gemm(wself[i], outwT[i], fin(pBase[i] + P_OUTB), delta[i], nullptr,
         Mrows[i], F[i], F[i], 0);
  // top-down contributions: project at half res, bilinear-upsample-add
  gemm(wtd[0], outwT[0] + (size_t)128 * 128, fin(pBase[0] + P_OUTB) + 128,
       tdtmp[0], nullptr, Mrows[0] / 4, 128, 128, 0);
  {
    int tot = Mrows[0] * F[0];
    upsample_add_kernel<<<(tot + 255) / 256, 256, 0, stream>>>(
        delta[0], tdtmp[0], 128, 128, 64, 64, 128, 63.f / 127.f, 63.f / 127.f, tot);
  }
  gemm(wtd[1], outwT[1] + (size_t)256 * 256, fin(pBase[1] + P_OUTB) + 256,
       tdtmp[1], nullptr, Mrows[1] / 4, 256, 256, 0);
  {
    int tot = Mrows[1] * F[1];
    upsample_add_kernel<<<(tot + 255) / 256, 256, 0, stream>>>(
        delta[1], tdtmp[1], 64, 64, 32, 32, 256, 31.f / 63.f, 31.f / 63.f, tot);
  }
  // bottom-up contributions: accumulate into deltas (ADD mode)
  gemm(wbu[0], outwT[1] + (size_t)2 * 256 * 256, fin(pBase[1] + P_OUTB) + 512,
       delta[1], nullptr, Mrows[1], 256, 256, 1);
  gemm(wbu[1], outwT[2] + (size_t)2 * 512 * 512, fin(pBase[2] + P_OUTB) + 1024,
       delta[2], nullptr, Mrows[2], 512, 512, 1);

  // ---- residual + LN + FFN + LN -----------------------------------------
  const size_t outOff[3] = {0, 4194304, 6291456};
  float* dout = (float*)d_out;
  for (int i = 0; i < 3; ++i) {
    int rows = Mrows[i], f = F[i];
    dim3 lg((rows + 7) / 8), lb(256);  // 8 wave32 rows per block
    ln_add_kernel<<<lg, lb, 0, stream>>>(
        fin(fIdx[i]), delta[i], fin(pBase[i] + P_LN1G), fin(pBase[i] + P_LN1B),
        xlnf[i], xlnb[i], rows, f);
    gemm(xlnb[i], w1T[i], fin(pBase[i] + P_B1), nullptr, hffb[i], rows, 4 * f, f, 2);
    gemm(hffb[i], w2T[i], fin(pBase[i] + P_B2), y2[i], nullptr, rows, f, 4 * f, 0);
    ln_add_kernel<<<lg, lb, 0, stream>>>(
        xlnf[i], y2[i], fin(pBase[i] + P_LN2G), fin(pBase[i] + P_LN2B),
        dout + outOff[i], nullptr, rows, f);
  }
}
